// PairwiseFeatureConstructor_66984309948704
// MI455X (gfx1250) — compile-verified, matrix operand order checked by reading, not run on hardware
//
#include <hip/hip_runtime.h>
#include <hip/hip_bf16.h>
#include <cstdint>

#define L  768
#define F  128
#define P  32
#define LL (L * L)

typedef __attribute__((ext_vector_type(4)))  _Float16 v4h;
typedef __attribute__((ext_vector_type(8)))  _Float16 v8h;
typedef __attribute__((ext_vector_type(16))) _Float16 v16h;
typedef __attribute__((ext_vector_type(8)))  float    v8f;

union Frag16 { v16h f; v8h h[2]; };

__device__ __forceinline__ float sigmoidf_(float x) { return 1.0f / (1.0f + __expf(-x)); }

// CDNA5 16-bit A/B fragment (16x32 / 32x16): lane m (0..15) holds K{0..7,16..23},
// lane m+16 holds K{8..15,24..31} of row/column m -> two b128 loads per lane.
__device__ __forceinline__ v16h load_frag(const _Float16* base, int rowStride, int m, int kh) {
  Frag16 fr;
  const _Float16* p = base + (size_t)m * rowStride + kh * 8;
  fr.h[0] = *(const v8h*)p;
  fr.h[1] = *(const v8h*)(p + 16);
  return fr.f;
}

// ---------------------------------------------------------------------------
// Kernel 1: LayerNorm over F, emit f16 x-hat. One wave32 per row.
// ---------------------------------------------------------------------------
__global__ __launch_bounds__(256) void k_layernorm(const float* __restrict__ feat,
                                                   const float* __restrict__ nw,
                                                   const float* __restrict__ nb,
                                                   _Float16* __restrict__ xh) {
  int tid = threadIdx.x, wave = tid >> 5, lane = tid & 31;
  size_t row = (size_t)blockIdx.x * 8 + wave;
  const float4 v = *(const float4*)(feat + row * F + lane * 4);
  float s = v.x + v.y + v.z + v.w;
  float q = v.x * v.x + v.y * v.y + v.z * v.z + v.w * v.w;
#pragma unroll
  for (int off = 16; off; off >>= 1) {
    s += __shfl_xor(s, off, 32);
    q += __shfl_xor(q, off, 32);
  }
  float mu = s * (1.0f / F);
  float rs = rsqrtf(q * (1.0f / F) - mu * mu + 1e-5f);
  const float4 w4 = *(const float4*)(nw + lane * 4);
  const float4 b4 = *(const float4*)(nb + lane * 4);
  v4h o;
  o.x = (_Float16)((v.x - mu) * rs * w4.x + b4.x);
  o.y = (_Float16)((v.y - mu) * rs * w4.y + b4.y);
  o.z = (_Float16)((v.z - mu) * rs * w4.z + b4.z);
  o.w = (_Float16)((v.w - mu) * rs * w4.w + b4.w);
  *(v4h*)(xh + row * F + lane * 4) = o;
}

// ---------------------------------------------------------------------------
// Kernel 2: fused WMMA GEMM  xh[LL,128] @ [aW;agW;gW]^T  (N = 32+32+128 = 192)
//   cols 0..31  -> lin, cols 32..63 -> sig-gate  => proj (f16, channel-major)
//   cols 64..191 -> sigmoid gate written into d_out (reused as gate buffer)
// ---------------------------------------------------------------------------
#define K2S 136  // padded LDS row stride (halves) -> conflict-free ds_load_b128
__global__ __launch_bounds__(256) void k_proj_gate(const _Float16* __restrict__ xh,
                                                   const float* __restrict__ aW,
                                                   const float* __restrict__ ab,
                                                   const float* __restrict__ agW,
                                                   const float* __restrict__ agb,
                                                   const float* __restrict__ gW,
                                                   const float* __restrict__ gb,
                                                   _Float16* __restrict__ projT,
                                                   float* __restrict__ gate_out) {
  __shared__ _Float16 Wc[192 * K2S];
  __shared__ float    bias[192];
  int tid = threadIdx.x;
  for (int idx = tid; idx < 192 * 128; idx += 256) {
    int r = idx >> 7, col = idx & 127;
    float v = (r < 32) ? aW[r * 128 + col]
            : (r < 64) ? agW[(r - 32) * 128 + col]
                       : gW[(r - 64) * 128 + col];
    Wc[r * K2S + col] = (_Float16)v;
  }
  if (tid < 192)
    bias[tid] = (tid < 32) ? ab[tid] : (tid < 64) ? agb[tid - 32] : gb[tid - 64];
  __syncthreads();

  int wave = tid >> 5, lane = tid & 31, m = lane & 15, kh = lane >> 4;
  size_t rowbase = (size_t)blockIdx.x * 128 + wave * 16;

  v8f acc[12];
#pragma unroll
  for (int t = 0; t < 12; ++t) acc[t] = v8f{};

#pragma unroll
  for (int kt = 0; kt < 4; ++kt) {
    v16h afr = load_frag(xh + rowbase * F + kt * 32, F, m, kh);
#pragma unroll
    for (int t = 0; t < 12; ++t) {
      v16h bfr = load_frag(Wc + t * 16 * K2S + kt * 32, K2S, m, kh);
      acc[t] = __builtin_amdgcn_wmma_f32_16x16x32_f16(false, afr, false, bfr,
                                                      (short)0, acc[t], false, false);
    }
  }

  // proj epilogue: lin * sigmoid(gate), 8 consecutive rows -> one b128 store
#pragma unroll
  for (int t = 0; t < 2; ++t) {
    int c = t * 16 + m;
    float bl = bias[c], bg = bias[32 + c];
    v8h pk;
#pragma unroll
    for (int k = 0; k < 8; ++k) {
      float lin = acc[t][k] + bl;
      float g   = sigmoidf_(acc[t + 2][k] + bg);
      pk[k] = (_Float16)(lin * g);
    }
    *(v8h*)(projT + (size_t)c * LL + rowbase + kh * 8) = pk;
  }
  // gate epilogue -> d_out (fp32)
#pragma unroll
  for (int t = 4; t < 12; ++t) {
    int f = (t - 4) * 16 + m;
    float bf = bias[64 + f];
#pragma unroll
    for (int k = 0; k < 8; ++k)
      gate_out[(rowbase + k + kh * 8) * F + f] = sigmoidf_(acc[t][k] + bf);
  }
}

// ---------------------------------------------------------------------------
// Kernel 3: per-channel gram tiles (WMMA) + LN over P + @oW^T (WMMA) + gate.
// One workgroup = 32x32 (i,j) tile, all 32 channels. Dynamic LDS ~156 KB.
// ---------------------------------------------------------------------------
#define PAIRS 1040  // per-channel pair-tile stride (halves)
#define PLNS  40    // padded K=32 row stride (halves)
__global__ __launch_bounds__(256) void k_einsum_out(const _Float16* __restrict__ projT,
                                                    const float* __restrict__ onw,
                                                    const float* __restrict__ onb,
                                                    const float* __restrict__ oW,
                                                    const float* __restrict__ ob,
                                                    float* __restrict__ out) {
  extern __shared__ char smem[];
  _Float16* pairb = (_Float16*)smem;            // [32][PAIRS]
  _Float16* pln   = pairb + 32 * PAIRS;         // [1024][PLNS]
  _Float16* oWh   = pln + 1024 * PLNS;          // [128][PLNS]
  float* sob  = (float*)(oWh + 128 * PLNS);     // [128]
  float* sonw = sob + 128;                      // [32]
  float* sonb = sonw + 32;                      // [32]

  int tid = threadIdx.x, wave = tid >> 5, lane = tid & 31, m = lane & 15, kh = lane >> 4;
  int i0 = blockIdx.x * 32, j0 = blockIdx.y * 32;

  for (int idx = tid; idx < 128 * 32; idx += 256)
    oWh[(idx >> 5) * PLNS + (idx & 31)] = (_Float16)oW[idx];
  if (tid < 128) sob[tid] = ob[tid];
  if (tid < 32) { sonw[tid] = onw[tid]; sonb[tid] = onb[tid]; }
  __syncthreads();

  // Phase 1: pair[c] tile = A_i(c) @ A_j(c)^T, 4 channels per wave
  for (int cc = 0; cc < 4; ++cc) {
    int c = wave * 4 + cc;
    const _Float16* baseA = projT + (size_t)c * LL + (size_t)i0 * L;
    const _Float16* baseB = projT + (size_t)c * LL + (size_t)j0 * L;
    v8f acc00 = v8f{}, acc01 = v8f{}, acc10 = v8f{}, acc11 = v8f{};
    for (int kt = 0; kt < 24; ++kt) {
      if (kt < 23) {  // L2 prefetch hints for the next k-slab
        __builtin_prefetch(baseA + (kt + 1) * 32, 0, 0);
        __builtin_prefetch(baseB + (kt + 1) * 32, 0, 0);
      }
      v16h a0 = load_frag(baseA + kt * 32, L, m, kh);
      v16h a1 = load_frag(baseA + 16 * L + kt * 32, L, m, kh);
      v16h b0 = load_frag(baseB + kt * 32, L, m, kh);
      v16h b1 = load_frag(baseB + 16 * L + kt * 32, L, m, kh);
      acc00 = __builtin_amdgcn_wmma_f32_16x16x32_f16(false, a0, false, b0, (short)0, acc00, false, false);
      acc01 = __builtin_amdgcn_wmma_f32_16x16x32_f16(false, a0, false, b1, (short)0, acc01, false, false);
      acc10 = __builtin_amdgcn_wmma_f32_16x16x32_f16(false, a1, false, b0, (short)0, acc10, false, false);
      acc11 = __builtin_amdgcn_wmma_f32_16x16x32_f16(false, a1, false, b1, (short)0, acc11, false, false);
    }
#pragma unroll
    for (int k = 0; k < 8; ++k) {
      int iiA = k + 8 * kh, iiB = 16 + k + 8 * kh;
      pairb[c * PAIRS + iiA * 32 + m]        = (_Float16)acc00[k];
      pairb[c * PAIRS + iiA * 32 + 16 + m]   = (_Float16)acc01[k];
      pairb[c * PAIRS + iiB * 32 + m]        = (_Float16)acc10[k];
      pairb[c * PAIRS + iiB * 32 + 16 + m]   = (_Float16)acc11[k];
    }
  }
  __syncthreads();

  // Phase 2: LayerNorm across the 32 channels at each (ii,jj) position
  for (int t = 0; t < 4; ++t) {
    int pos = tid + t * 256;
    float vals[32], s = 0.f, q = 0.f;
#pragma unroll
    for (int c = 0; c < 32; ++c) {
      float v = (float)pairb[c * PAIRS + pos];
      vals[c] = v; s += v; q += v * v;
    }
    float mu = s * (1.0f / 32.0f);
    float rs = rsqrtf(q * (1.0f / 32.0f) - mu * mu + 1e-5f);
#pragma unroll
    for (int c = 0; c < 32; ++c)
      pln[pos * PLNS + c] = (_Float16)((vals[c] - mu) * rs * sonw[c] + sonb[c]);
  }
  __syncthreads();

  // Phase 3: out = (pln @ oW^T + ob) * gate  (K=32 -> single WMMA step)
  for (int mt = 0; mt < 8; ++mt) {
    int mtile = wave * 8 + mt;
    v16h afr = load_frag(pln + mtile * 16 * PLNS, PLNS, m, kh);
    for (int nt = 0; nt < 8; ++nt) {
      v16h bfr = load_frag(oWh + nt * 16 * PLNS, PLNS, m, kh);
      v8f acc = v8f{};
      acc = __builtin_amdgcn_wmma_f32_16x16x32_f16(false, afr, false, bfr,
                                                   (short)0, acc, false, false);
      int f = nt * 16 + m;
      float bo = sob[f];
#pragma unroll
      for (int k = 0; k < 8; ++k) {
        int pos = mtile * 16 + k + 8 * kh;
        int ii = pos >> 5, jj = pos & 31;
        size_t gi = ((size_t)(i0 + ii) * L + (j0 + jj)) * F + f;
        out[gi] = (acc[k] + bo) * out[gi];  // gate was staged in d_out
      }
    }
  }
}

// ---------------------------------------------------------------------------
extern "C" void kernel_launch(void* const* d_in, const int* in_sizes, int n_in,
                              void* d_out, int out_size, void* d_ws, size_t ws_size,
                              hipStream_t stream) {
  (void)in_sizes; (void)n_in; (void)out_size; (void)ws_size;
  const float* feat = (const float*)d_in[0];
  const float* nw   = (const float*)d_in[1];
  const float* nb   = (const float*)d_in[2];
  const float* onw  = (const float*)d_in[3];
  const float* onb  = (const float*)d_in[4];
  const float* aW   = (const float*)d_in[5];
  const float* ab   = (const float*)d_in[6];
  const float* agW  = (const float*)d_in[7];
  const float* agb  = (const float*)d_in[8];
  const float* oW   = (const float*)d_in[9];
  const float* ob   = (const float*)d_in[10];
  const float* gW   = (const float*)d_in[11];
  const float* gb   = (const float*)d_in[12];

  _Float16* xh    = (_Float16*)d_ws;                 // 151 MB
  _Float16* projT = xh + (size_t)LL * F;             // +37.7 MB (channel-major)
  float*    out   = (float*)d_out;

  k_layernorm<<<LL / 8, 256, 0, stream>>>(feat, nw, nb, xh);
  k_proj_gate<<<LL / 128, 256, 0, stream>>>(xh, aW, ab, agW, agb, gW, gb, projT, out);

  size_t smem = (size_t)(32 * PAIRS + 1024 * PLNS + 128 * PLNS) * sizeof(_Float16)
              + 192 * sizeof(float);  // ~156 KB < 320 KB/WGP
  k_einsum_out<<<dim3(24, 24), 256, smem, stream>>>(projT, onw, onb, oW, ob, out);
}
